// CausalEncoder_22814866276516
// MI455X (gfx1250) — matrix operand order repaired and checked by reading, not compiled, not run on hardware
//
#include <hip/hip_runtime.h>
#include <hip/hip_bf16.h>
#include <math.h>

// ---------------- config (Qwen2-0.5B style, 4 layers) ----------------
#define L_    4
#define H_    14
#define KV_   2
#define D_    64
#define HID_  896
#define FF_   4864
#define NV_   784
#define NQ_   392
#define T_    1176
#define B_    4
#define TPAD_ 1184          // T padded to multiple of 16 for WMMA tiling
#define ROWS_ (B_ * T_)     // 4704
#define EPS_  1e-6f

typedef __bf16 bf16_t;
typedef __attribute__((ext_vector_type(16))) __bf16 v16bf;
typedef __attribute__((ext_vector_type(8)))  __bf16 bf16x8;
typedef __attribute__((ext_vector_type(8)))  float  v8f;

// D = A(16x32 bf16) * B(32x16 bf16) + C(16x16 f32)
static __device__ inline v8f wmma_bf16(v16bf a, v16bf b, v8f c) {
  return __builtin_amdgcn_wmma_f32_16x16x32_bf16(
      false, a, false, b, (short)0, c, false, false);
}

// A-fragment: lane<16 -> row=lane, K chunks {hi8..hi8+7, 16+hi8..23+hi8}, hi8 = (lane&16)?8:0
static __device__ inline v16bf load_a_frag(const bf16_t* rowp, int hi8) {
  bf16x8 lo = *(const bf16x8*)(rowp + hi8);
  bf16x8 hi = *(const bf16x8*)(rowp + 16 + hi8);
  return __builtin_shufflevector(lo, hi, 0, 1, 2, 3, 4, 5, 6, 7,
                                 8, 9, 10, 11, 12, 13, 14, 15);
}
// B-fragment: same K chunks as A: {hi8..hi8+7, 16+hi8..23+hi8}, hi8 = hi16/2
static __device__ inline v16bf load_b_frag(const bf16_t* colp, int hi16) {
  bf16x8 lo = *(const bf16x8*)(colp + (hi16 >> 1));
  bf16x8 hi = *(const bf16x8*)(colp + 16 + (hi16 >> 1));
  return __builtin_shufflevector(lo, hi, 0, 1, 2, 3, 4, 5, 6, 7,
                                 8, 9, 10, 11, 12, 13, 14, 15);
}

// ---- async global->LDS staging (gfx1250 GLOBAL_LOAD_ASYNC_TO_LDS_B128) ----
static __device__ inline void async_copy16(const bf16_t* g, bf16_t* l) {
#if defined(__gfx1250__)
  unsigned lds_off =
      (unsigned)(unsigned long long)(__attribute__((address_space(3))) bf16_t*)l;
  asm volatile("global_load_async_to_lds_b128 %0, %1, off"
               :: "v"(lds_off), "v"((unsigned long long)g)
               : "memory");
#else
  *(float4*)l = *(const float4*)g;
#endif
}
static __device__ inline void wait_async() {
#if __has_builtin(__builtin_amdgcn_s_wait_asynccnt)
  __builtin_amdgcn_s_wait_asynccnt(0);
#elif defined(__gfx1250__)
  asm volatile("s_wait_asynccnt 0" ::: "memory");
#endif
}

// ---------------- concat: x = [visual ; broadcast(queries)] ----------------
__global__ __launch_bounds__(256) void concat_kernel(
    const float* __restrict__ vis, const float* __restrict__ qry,
    float* __restrict__ x) {
  size_t i = (size_t)blockIdx.x * 256 + threadIdx.x;
  size_t total = (size_t)ROWS_ * HID_;
  if (i >= total) return;
  int c = (int)(i % HID_);
  size_t r = i / HID_;
  int t = (int)(r % T_);
  int b = (int)(r / T_);
  x[i] = (t < NV_) ? vis[((size_t)b * NV_ + t) * HID_ + c]
                   : qry[(size_t)(t - NV_) * HID_ + c];
}

// ---------------- weight transpose + f32->bf16: W[K][N] -> Wt[N][K] ----------------
__global__ __launch_bounds__(256) void transpose_bf16_kernel(
    const float* __restrict__ src, bf16_t* __restrict__ dst, int K, int N) {
  __shared__ float tile[32][33];
  int kb = blockIdx.y * 32, nb = blockIdx.x * 32;
  int tx = threadIdx.x, ty = threadIdx.y;   // (32, 8)
#pragma unroll
  for (int i = 0; i < 32; i += 8)
    tile[ty + i][tx] = src[(size_t)(kb + ty + i) * N + nb + tx];
  __syncthreads();
#pragma unroll
  for (int i = 0; i < 32; i += 8)
    dst[(size_t)(nb + ty + i) * K + kb + tx] = (bf16_t)tile[tx][ty + i];
}

// ---------------- RMSNorm: one block per row ----------------
__global__ __launch_bounds__(256) void rmsnorm_kernel(
    const float* __restrict__ x, const float* __restrict__ w,
    bf16_t* __restrict__ out_bf, float* __restrict__ out_f32) {
  __shared__ float red[256];
  int row = blockIdx.x;
  int tid = threadIdx.x;
  const float* xr = x + (size_t)row * HID_;
  float ss = 0.f;
  for (int i = tid; i < HID_; i += 256) { float v = xr[i]; ss += v * v; }
  red[tid] = ss;
  __syncthreads();
  for (int s = 128; s > 0; s >>= 1) {
    if (tid < s) red[tid] += red[tid + s];
    __syncthreads();
  }
  float rstd = rsqrtf(red[0] / (float)HID_ + EPS_);
  int t = row % T_;
  int b = row / T_;
  for (int i = tid; i < HID_; i += 256) {
    float v = xr[i] * rstd * w[i];
    if (out_bf) out_bf[(size_t)row * HID_ + i] = (bf16_t)v;
    if (out_f32 && t >= NV_)
      out_f32[((size_t)b * NQ_ + (t - NV_)) * HID_ + i] = v;
  }
}

// ---------------- WMMA GEMM: C[M,N] = A_bf16[M,K] @ Wt_bf16[N][K]^T ----------------
// mode 0: C = acc (+bias)   mode 1: C += acc   mode 2: Cbf = silu(aux)*acc
#define BM 64
#define BN 128
#define BKK 64
__global__ __launch_bounds__(256) void gemm_kernel(
    const bf16_t* __restrict__ A, const bf16_t* __restrict__ Wt,
    const float* __restrict__ bias, float* __restrict__ C,
    const float* __restrict__ aux, bf16_t* __restrict__ Cbf,
    int M, int N, int K, int mode) {
  __shared__ __align__(128) bf16_t As[BM][BKK];   // 8 KB
  __shared__ __align__(128) bf16_t Ws[BN][BKK];   // 16 KB (column tile of W == row of Wt)
  int tid  = threadIdx.x;
  int lane = tid & 31;
  int wid  = tid >> 5;            // 0..7
  int m_blk = blockIdx.y * BM;
  int n_blk = blockIdx.x * BN;
  int wm = (wid >> 1) * 16;
  int wn = (wid & 1) * 64;
  int hi8  = (lane & 16) ? 8 : 0;
  int hi16 = (lane & 16) ? 16 : 0;

  v8f acc[4] = {};

  for (int k0 = 0; k0 < K; k0 += BKK) {
    // stage A tile: 64 x 64 bf16 = 512 x 16B chunks
#pragma unroll
    for (int i = 0; i < 2; ++i) {
      int id = tid + i * 256;
      int r = id >> 3, c = (id & 7) * 8;
      int gr = m_blk + r; if (gr >= M) gr = M - 1;     // clamp; bad rows never stored
      async_copy16(A + (size_t)gr * K + k0 + c, &As[r][c]);
    }
    // stage Wt tile: 128 x 64 bf16 = 1024 x 16B chunks
#pragma unroll
    for (int i = 0; i < 4; ++i) {
      int id = tid + i * 256;
      int n = id >> 3, c = (id & 7) * 8;
      const bf16_t* src = Wt + (size_t)(n_blk + n) * K + k0 + c;
      async_copy16(src, &Ws[n][c]);
      if (i == 0 && k0 + BKK < K)
        __builtin_prefetch(src + BKK, 0, 1);           // global_prefetch_b8
    }
    wait_async();
    __syncthreads();

#pragma unroll
    for (int kk = 0; kk < BKK; kk += 32) {
      v16bf af = load_a_frag(&As[wm + (lane & 15)][kk], hi8);
#pragma unroll
      for (int j = 0; j < 4; ++j) {
        v16bf bfr = load_b_frag(&Ws[wn + j * 16 + (lane & 15)][kk], hi16);
        acc[j] = wmma_bf16(af, bfr, acc[j]);
      }
    }
    __syncthreads();
  }

  int nloc  = lane & 15;
  int mbase = (lane & 16) ? 8 : 0;
#pragma unroll
  for (int j = 0; j < 4; ++j) {
    int gn = n_blk + wn + j * 16 + nloc;
#pragma unroll
    for (int r = 0; r < 8; ++r) {
      int gm = m_blk + wm + mbase + r;
      if (gm >= M) continue;
      float v = acc[j][r];
      size_t idx = (size_t)gm * N + gn;
      if (mode == 0)      C[idx] = v + (bias ? bias[gn] : 0.f);
      else if (mode == 1) C[idx] += v;
      else {
        float g  = aux[idx];
        float sg = g / (1.f + __expf(-g));
        Cbf[idx] = (bf16_t)(sg * v);
      }
    }
  }
}

// ---------------- RoPE + repack Q/K to [B, heads, TPAD, D] bf16 ----------------
__global__ __launch_bounds__(256) void rope_pack_kernel(
    const float* __restrict__ in, bf16_t* __restrict__ out, int heads) {
  size_t i = (size_t)blockIdx.x * 256 + threadIdx.x;
  size_t total = (size_t)B_ * heads * TPAD_ * D_;
  if (i >= total) return;
  int d = (int)(i & (D_ - 1));
  size_t r = i >> 6;
  int t = (int)(r % TPAD_); r /= TPAD_;
  int h = (int)(r % heads);
  int b = (int)(r / heads);
  float val = 0.f;
  if (t < T_) {
    const float* src = in + ((size_t)(b * T_ + t) * heads + h) * D_;
    float v = src[d];
    int fi = d & 31;
    float inv_f = __expf(-(float)(2 * fi) / (float)D_ * 13.815511f); // ln(1e6)
    float ang = (float)t * inv_f;
    float c = __cosf(ang), s = __sinf(ang);
    float other = (d < 32) ? -src[d + 32] : src[d - 32];
    val = v * c + other * s;
  }
  out[i] = (bf16_t)val;
}

// ---------------- V repack transposed: [B][KV][D][TPAD] bf16 ----------------
__global__ __launch_bounds__(256) void vpack_kernel(
    const float* __restrict__ in, bf16_t* __restrict__ out) {
  size_t i = (size_t)blockIdx.x * 256 + threadIdx.x;
  size_t total = (size_t)B_ * KV_ * D_ * TPAD_;
  if (i >= total) return;
  int t = (int)(i % TPAD_);
  size_t r = i / TPAD_;
  int d = (int)(r % D_); r /= D_;
  int h = (int)(r % KV_);
  int b = (int)(r / KV_);
  float val = (t < T_) ? in[((size_t)(b * T_ + t) * KV_ + h) * D_ + d] : 0.f;
  out[i] = (bf16_t)val;
}

// ---------------- fused attention: one wave per (b, head, 16-query tile) ----------------
__global__ __launch_bounds__(32) void attention_kernel(
    const bf16_t* __restrict__ qbf, const bf16_t* __restrict__ kbf,
    const bf16_t* __restrict__ vbf, bf16_t* __restrict__ out_bf) {
  extern __shared__ char smem_raw[];
  float (*sc)[TPAD_] = (float(*)[TPAD_])smem_raw;   // 16 x 1184 f32 = 75776 B

  int lane = threadIdx.x;
  int qt = blockIdx.x;
  int h  = blockIdx.y;
  int b  = blockIdx.z;
  int kvh = h / (H_ / KV_);
  int hi8  = (lane & 16) ? 8 : 0;
  int hi16 = (lane & 16) ? 16 : 0;
  int mbase = (lane & 16) ? 8 : 0;

  const bf16_t* Q  = qbf + ((size_t)(b * H_  + h)   * TPAD_) * D_;
  const bf16_t* Km = kbf + ((size_t)(b * KV_ + kvh) * TPAD_) * D_;
  const bf16_t* Vt = vbf + ((size_t)(b * KV_ + kvh) * D_) * TPAD_;

  int qrow = qt * 16 + (lane & 15);
  v16bf qa[2];
#pragma unroll
  for (int half = 0; half < 2; ++half)
    qa[half] = load_a_frag(Q + (size_t)qrow * D_ + half * 32, hi8);

  const float scale = 0.125f;   // 1/sqrt(64)

  // S = Q K^T with custom mask -> LDS
  for (int kt = 0; kt < TPAD_ / 16; ++kt) {
    v8f s = {};
    int kcol = kt * 16 + (lane & 15);
#pragma unroll
    for (int half = 0; half < 2; ++half) {
      v16bf kb = load_b_frag(Km + (size_t)kcol * D_ + half * 32, hi16);
      s = wmma_bf16(qa[half], kb, s);
    }
#pragma unroll
    for (int r = 0; r < 8; ++r) {
      int qi = qt * 16 + mbase + r;
      bool ok = (qi < T_) && (kcol < T_) &&
                (((qi < NV_) && (kcol < NV_)) ||
                 ((qi >= NV_) && (kcol >= NV_) && (kcol <= qi)));
      sc[mbase + r][kcol] = ok ? s[r] * scale : -1e30f;
    }
  }
  __syncthreads();

  // row softmax: 2 lanes per row, halves combined via lane shuffle
  {
    int rrow = lane & 15;
    int c0 = (lane >> 4) * (TPAD_ / 2);
    float mx = -1e30f;
    for (int j = c0; j < c0 + TPAD_ / 2; ++j) mx = fmaxf(mx, sc[rrow][j]);
    mx = fmaxf(mx, __shfl_xor(mx, 16, 32));
    float sum = 0.f;
    for (int j = c0; j < c0 + TPAD_ / 2; ++j) {
      float e = __expf(sc[rrow][j] - mx);
      sc[rrow][j] = e;
      sum += e;
    }
    sum += __shfl_xor(sum, 16, 32);
    float inv = 1.f / sum;
    for (int j = c0; j < c0 + TPAD_ / 2; ++j) sc[rrow][j] *= inv;
  }
  __syncthreads();

  // O = P V  (V stored transposed: Vt[d][t])
  v8f o[4] = {};
  int prow = lane & 15;
  for (int kc = 0; kc < TPAD_ / 32; ++kc) {
    v8f p0 = *(const v8f*)&sc[prow][kc * 32 + hi8];
    v8f p1 = *(const v8f*)&sc[prow][kc * 32 + 16 + hi8];
    v16bf pa;
#pragma unroll
    for (int e = 0; e < 8; ++e) { pa[e] = (bf16_t)p0[e]; pa[e + 8] = (bf16_t)p1[e]; }
#pragma unroll
    for (int j = 0; j < 4; ++j) {
      int vcol = j * 16 + (lane & 15);
      v16bf vb = load_b_frag(Vt + (size_t)vcol * TPAD_ + kc * 32, hi16);
      o[j] = wmma_bf16(pa, vb, o[j]);
    }
  }

#pragma unroll
  for (int j = 0; j < 4; ++j) {
    int col = h * D_ + j * 16 + (lane & 15);
#pragma unroll
    for (int r = 0; r < 8; ++r) {
      int t = qt * 16 + mbase + r;
      if (t < T_)
        out_bf[((size_t)(b * T_ + t)) * (H_ * D_) + col] = (bf16_t)o[j][r];
    }
  }
}

// ---------------- host orchestration ----------------
static inline size_t align256(size_t x) { return (x + 255) & ~(size_t)255; }

extern "C" void kernel_launch(void* const* d_in, const int* in_sizes, int n_in,
                              void* d_out, int out_size, void* d_ws, size_t ws_size,
                              hipStream_t stream) {
  (void)in_sizes; (void)n_in; (void)out_size; (void)ws_size;
  const float* vis  = (const float*)d_in[0];
  const float* qry  = (const float*)d_in[1];
  const float* wq   = (const float*)d_in[2];
  const float* bq   = (const float*)d_in[3];
  const float* wk   = (const float*)d_in[4];
  const float* bk   = (const float*)d_in[5];
  const float* wv   = (const float*)d_in[6];
  const float* bv   = (const float*)d_in[7];
  const float* wo   = (const float*)d_in[8];
  const float* ln1  = (const float*)d_in[9];
  const float* ln2  = (const float*)d_in[10];
  const float* wg   = (const float*)d_in[11];
  const float* wu   = (const float*)d_in[12];
  const float* wd   = (const float*)d_in[13];
  const float* lnf  = (const float*)d_in[14];
  float* out = (float*)d_out;

  char* ws = (char*)d_ws;
  size_t off = 0;
  auto alloc = [&](size_t bytes) { char* p = ws + off; off += align256(bytes); return p; };
  float*  x      = (float*) alloc((size_t)ROWS_ * HID_ * 4);
  bf16_t* hbf    = (bf16_t*)alloc((size_t)ROWS_ * HID_ * 2);
  float*  qf     = (float*) alloc((size_t)ROWS_ * H_ * D_ * 4);
  float*  kf     = (float*) alloc((size_t)ROWS_ * KV_ * D_ * 4);
  float*  vf     = (float*) alloc((size_t)ROWS_ * KV_ * D_ * 4);
  bf16_t* qbf    = (bf16_t*)alloc((size_t)B_ * H_  * TPAD_ * D_ * 2);
  bf16_t* kbf    = (bf16_t*)alloc((size_t)B_ * KV_ * TPAD_ * D_ * 2);
  bf16_t* vbf    = (bf16_t*)alloc((size_t)B_ * KV_ * D_ * TPAD_ * 2);
  bf16_t* attnbf = (bf16_t*)alloc((size_t)ROWS_ * H_ * D_ * 2);
  float*  gbuf   = (float*) alloc((size_t)ROWS_ * FF_ * 4);
  bf16_t* actbf  = (bf16_t*)alloc((size_t)ROWS_ * FF_ * 2);
  // transposed bf16 weights
  bf16_t* wqT = (bf16_t*)alloc((size_t)L_ * HID_ * H_ * D_ * 2);
  bf16_t* wkT = (bf16_t*)alloc((size_t)L_ * HID_ * KV_ * D_ * 2);
  bf16_t* wvT = (bf16_t*)alloc((size_t)L_ * HID_ * KV_ * D_ * 2);
  bf16_t* woT = (bf16_t*)alloc((size_t)L_ * H_ * D_ * HID_ * 2);
  bf16_t* wgT = (bf16_t*)alloc((size_t)L_ * HID_ * FF_ * 2);
  bf16_t* wuT = (bf16_t*)alloc((size_t)L_ * HID_ * FF_ * 2);
  bf16_t* wdT = (bf16_t*)alloc((size_t)L_ * FF_ * HID_ * 2);

  const size_t attn_lds = 16 * TPAD_ * sizeof(float);
  (void)hipFuncSetAttribute((const void*)attention_kernel,
                            hipFuncAttributeMaxDynamicSharedMemorySize, (int)attn_lds);

  // one-time (per launch) weight transpose/convert
  auto xpose = [&](const float* s, bf16_t* d, int K, int N) {
    dim3 g(N / 32, K / 32), blk(32, 8);
    transpose_bf16_kernel<<<g, blk, 0, stream>>>(s, d, K, N);
  };
  for (int l = 0; l < L_; ++l) {
    xpose(wq + (size_t)l * HID_ * H_ * D_,  wqT + (size_t)l * HID_ * H_ * D_,  HID_, H_ * D_);
    xpose(wk + (size_t)l * HID_ * KV_ * D_, wkT + (size_t)l * HID_ * KV_ * D_, HID_, KV_ * D_);
    xpose(wv + (size_t)l * HID_ * KV_ * D_, wvT + (size_t)l * HID_ * KV_ * D_, HID_, KV_ * D_);
    xpose(wo + (size_t)l * H_ * D_ * HID_,  woT + (size_t)l * H_ * D_ * HID_,  H_ * D_, HID_);
    xpose(wg + (size_t)l * HID_ * FF_,      wgT + (size_t)l * HID_ * FF_,      HID_, FF_);
    xpose(wu + (size_t)l * HID_ * FF_,      wuT + (size_t)l * HID_ * FF_,      HID_, FF_);
    xpose(wd + (size_t)l * FF_ * HID_,      wdT + (size_t)l * FF_ * HID_,      FF_, HID_);
  }

  {
    size_t total = (size_t)ROWS_ * HID_;
    concat_kernel<<<(unsigned)((total + 255) / 256), 256, 0, stream>>>(vis, qry, x);
  }

  dim3 gQ(HID_ / BN, (ROWS_ + BM - 1) / BM);
  dim3 gKV((KV_ * D_) / BN, (ROWS_ + BM - 1) / BM);
  dim3 gFF(FF_ / BN, (ROWS_ + BM - 1) / BM);
  dim3 gAtt(TPAD_ / 16, H_, B_);

  size_t ropeQ = (size_t)B_ * H_  * TPAD_ * D_;
  size_t ropeK = (size_t)B_ * KV_ * TPAD_ * D_;

  for (int l = 0; l < L_; ++l) {
    bf16_t* wqT_l = wqT + (size_t)l * HID_ * H_ * D_;
    bf16_t* wkT_l = wkT + (size_t)l * HID_ * KV_ * D_;
    bf16_t* wvT_l = wvT + (size_t)l * HID_ * KV_ * D_;
    bf16_t* woT_l = woT + (size_t)l * H_ * D_ * HID_;
    bf16_t* wgT_l = wgT + (size_t)l * HID_ * FF_;
    bf16_t* wuT_l = wuT + (size_t)l * HID_ * FF_;
    bf16_t* wdT_l = wdT + (size_t)l * FF_ * HID_;
    const float* bq_l = bq + (size_t)l * H_ * D_;
    const float* bk_l = bk + (size_t)l * KV_ * D_;
    const float* bv_l = bv + (size_t)l * KV_ * D_;

    rmsnorm_kernel<<<ROWS_, 256, 0, stream>>>(x, ln1 + (size_t)l * HID_, hbf, nullptr);
    gemm_kernel<<<gQ,  256, 0, stream>>>(hbf, wqT_l, bq_l, qf, nullptr, nullptr,
                                         ROWS_, H_ * D_, HID_, 0);
    gemm_kernel<<<gKV, 256, 0, stream>>>(hbf, wkT_l, bk_l, kf, nullptr, nullptr,
                                         ROWS_, KV_ * D_, HID_, 0);
    gemm_kernel<<<gKV, 256, 0, stream>>>(hbf, wvT_l, bv_l, vf, nullptr, nullptr,
                                         ROWS_, KV_ * D_, HID_, 0);
    rope_pack_kernel<<<(unsigned)((ropeQ + 255) / 256), 256, 0, stream>>>(qf, qbf, H_);
    rope_pack_kernel<<<(unsigned)((ropeK + 255) / 256), 256, 0, stream>>>(kf, kbf, KV_);
    vpack_kernel<<<(unsigned)((ropeK + 255) / 256), 256, 0, stream>>>(vf, vbf);
    attention_kernel<<<gAtt, 32, attn_lds, stream>>>(qbf, kbf, vbf, attnbf);
    gemm_kernel<<<gQ, 256, 0, stream>>>(attnbf, woT_l, nullptr, x, nullptr, nullptr,
                                        ROWS_, HID_, H_ * D_, 1);
    rmsnorm_kernel<<<ROWS_, 256, 0, stream>>>(x, ln2 + (size_t)l * HID_, hbf, nullptr);
    gemm_kernel<<<gFF, 256, 0, stream>>>(hbf, wgT_l, nullptr, gbuf, nullptr, nullptr,
                                         ROWS_, FF_, HID_, 0);
    gemm_kernel<<<gFF, 256, 0, stream>>>(hbf, wuT_l, nullptr, nullptr, gbuf, actbf,
                                         ROWS_, FF_, HID_, 2);
    gemm_kernel<<<gQ,  256, 0, stream>>>(actbf, wdT_l, nullptr, x, nullptr, nullptr,
                                         ROWS_, HID_, FF_, 1);
  }

  rmsnorm_kernel<<<ROWS_, 256, 0, stream>>>(x, lnf, nullptr, out);
}
